// VJEPA2Predictor_2869038153988
// MI455X (gfx1250) — matrix operand order repaired and checked by reading, not compiled, not run on hardware
//
#include <hip/hip_runtime.h>
#include <hip/hip_bf16.h>
#include <math.h>

// ---------------------------------------------------------------- constants
#define Bq      4
#define NSEQ    1568
#define NCTX    1176
#define NTGT    392
#define ENCD    1024
#define DMODEL  384
#define NHEAD   12
#define NLAYER  12
#define HD      32
#define MLPD    1536
#define NTOK    1568
#define TOTROWS (Bq * NTOK)   // 6272
#define ATT_SCALE 0.17677669529663687f   // 32^-0.5

typedef _Float16 half_t;
typedef __attribute__((ext_vector_type(16))) _Float16 v16h;
typedef __attribute__((ext_vector_type(8)))  _Float16 v8h;
typedef __attribute__((ext_vector_type(8)))  float    v8f;
typedef __attribute__((ext_vector_type(4)))  int      i4;

union F16x16 { v16h v; v8h h[2]; };

#define AS1 __attribute__((address_space(1)))
#define AS3 __attribute__((address_space(3)))

#if __has_builtin(__builtin_amdgcn_global_load_async_to_lds_b128)
#define HAVE_ASYNC_LDS 1
#endif

static __device__ __forceinline__ void lds_wait() {
#if __has_builtin(__builtin_amdgcn_s_wait_dscnt)
  __builtin_amdgcn_s_wait_dscnt(0);
#else
  asm volatile("s_wait_dscnt 0" ::: "memory");
#endif
}

static __device__ __forceinline__ void async_wait() {
#ifdef HAVE_ASYNC_LDS
#if __has_builtin(__builtin_amdgcn_s_wait_asynccnt)
  __builtin_amdgcn_s_wait_asynccnt(0);
#else
  asm volatile("s_wait_asynccnt 0" ::: "memory");
#endif
#endif
}

static __device__ __forceinline__ void async_copy16(const half_t* g, half_t* l) {
#ifdef HAVE_ASYNC_LDS
  __builtin_amdgcn_global_load_async_to_lds_b128((AS1 i4*)g, (AS3 i4*)l, 0, 0);
#else
  *(v8h*)l = *(const v8h*)g;
#endif
}

// ---------------------------------------------------------------- weight transpose f32(K,N) -> f16(N,K)
__global__ void transpose_to_f16_kernel(const float* __restrict__ src,
                                        half_t* __restrict__ dst, int K, int N) {
  size_t idx = (size_t)blockIdx.x * 256 + threadIdx.x;
  if (idx >= (size_t)K * N) return;
  int n = (int)(idx / K), k = (int)(idx % K);
  dst[idx] = (half_t)src[(size_t)k * N + n];
}

// ---------------------------------------------------------------- index builders
__global__ void build_rows_kernel(const int* __restrict__ cm, const int* __restrict__ tm,
                                  int* __restrict__ ctx_rows, int* __restrict__ tgt_rows) {
  int i = blockIdx.x * 256 + threadIdx.x;
  if (i < Bq * NCTX) ctx_rows[i] = (i / NCTX) * NTOK + cm[i];
  if (i < Bq * NTGT) tgt_rows[i] = (i / NTGT) * NTOK + tm[i];
}

__global__ void init_tgt_kernel(const int* __restrict__ tgt_rows,
                                const float* __restrict__ mask_token,
                                float* __restrict__ h) {
  int i = blockIdx.x;
  int r = tgt_rows[i];
  for (int c = threadIdx.x; c < DMODEL; c += 128)
    h[(size_t)r * DMODEL + c] = mask_token[c];
}

__global__ void gather_enc_kernel(const float* __restrict__ enc,
                                  const int* __restrict__ cm,
                                  half_t* __restrict__ Actx) {
  int i = blockIdx.x;
  int b = i / NCTX;
  const float* s = enc + ((size_t)b * NSEQ + cm[i]) * ENCD;
  half_t* d = Actx + (size_t)i * ENCD;
  for (int c = threadIdx.x; c < ENCD; c += 256) d[c] = (half_t)s[c];
}

__global__ void gather_f16_kernel(const half_t* __restrict__ xln,
                                  const int* __restrict__ rows,
                                  half_t* __restrict__ out) {
  int i = blockIdx.x;
  const half_t* s = xln + (size_t)rows[i] * DMODEL;
  half_t* d = out + (size_t)i * DMODEL;
  for (int c = threadIdx.x; c < DMODEL; c += 128) d[c] = s[c];
}

// ---------------------------------------------------------------- LayerNorm (wave per row, D=384)
__global__ __launch_bounds__(128)
void layernorm_kernel(const float* __restrict__ X, const float* __restrict__ g,
                      const float* __restrict__ be, half_t* __restrict__ out, int R) {
  int lane = threadIdx.x & 31, wave = threadIdx.x >> 5;
  int row = blockIdx.x * 4 + wave;
  if (row >= R) return;
  const float* x = X + (size_t)row * DMODEL;
  float v[12];
  float s = 0.f;
#pragma unroll
  for (int i = 0; i < 12; ++i) { v[i] = x[lane + 32 * i]; s += v[i]; }
  for (int d = 1; d < 32; d <<= 1) s += __shfl_xor(s, d, 32);
  float mean = s * (1.f / 384.f);
  float q = 0.f;
#pragma unroll
  for (int i = 0; i < 12; ++i) { float t = v[i] - mean; q += t * t; }
  for (int d = 1; d < 32; d <<= 1) q += __shfl_xor(q, d, 32);
  float rstd = rsqrtf(q * (1.f / 384.f) + 1e-6f);
  half_t* o = out + (size_t)row * DMODEL;
#pragma unroll
  for (int i = 0; i < 12; ++i) {
    int c = lane + 32 * i;
    o[c] = (half_t)((v[i] - mean) * rstd * g[c] + be[c]);
  }
}

// ---------------------------------------------------------------- RoPE + head pack
__global__ void rope_pack_kernel(const float* __restrict__ qf, const float* __restrict__ kf,
                                 const float* __restrict__ vf,
                                 half_t* __restrict__ Qh, half_t* __restrict__ Kh,
                                 half_t* __restrict__ Vt) {
  size_t idx = (size_t)blockIdx.x * 256 + threadIdx.x;
  if (idx >= (size_t)TOTROWS * DMODEL) return;
  int row = (int)(idx / DMODEL), d = (int)(idx % DMODEL);
  int b = row / NTOK, n = row % NTOK;
  int h = d >> 5, hd = d & 31;
  float xq = qf[idx], xk = kf[idx], xv = vf[idx];
  float oq = xq, ok = xk;
  if (hd < 30) {
    int j, p;
    if (hd < 10)      { j = hd;      p = n / 196; }
    else if (hd < 20) { j = hd - 10; p = (n % 196) / 14; }
    else              { j = hd - 20; p = (n % 196) % 14; }
    float freq = (float)p * powf(10000.f, -(float)(j % 5) * 0.2f);
    float c = cosf(freq), sn = sinf(freq);
    bool even = (hd & 1) == 0;
    float rq = even ? -qf[idx + 1] : qf[idx - 1];
    float rk = even ? -kf[idx + 1] : kf[idx - 1];
    oq = xq * c + rq * sn;
    ok = xk * c + rk * sn;
  }
  size_t po = ((size_t)(b * NHEAD + h) * NTOK + n) * HD + hd;
  Qh[po] = (half_t)oq;
  Kh[po] = (half_t)ok;
  Vt[((size_t)(b * NHEAD + h) * HD + hd) * NTOK + n] = (half_t)xv;
}

// ---------------------------------------------------------------- WMMA GEMM, LDS double-buffered
// C[M,N] = A[M,K](f16) * Bt[N,K](f16)^T  (+bias)(+GELU)(+residual) -> f32 or f16, optional row scatter.
// Block: 256 threads (8 waves). Tile 128(M) x 128(N), k-step 32.
// Tiles streamed into LDS with GLOBAL_LOAD_ASYNC_TO_LDS_B128 (double buffered).
__global__ __launch_bounds__(256)
void gemm_f16_kernel(const half_t* __restrict__ A, const half_t* __restrict__ Bt,
                     const float* __restrict__ bias, const float* __restrict__ res,
                     float* __restrict__ outF, half_t* __restrict__ outH,
                     const int* __restrict__ outRows, int M, int N, int K, int act) {
  __shared__ __align__(32) half_t As[2][128 * 32];
  __shared__ __align__(32) half_t Bs[2][128 * 32];
  const int tid = threadIdx.x;
  const int lane = tid & 31, wave = tid >> 5;
  const int m0 = blockIdx.y * 128;
  const int n0 = blockIdx.x * 128;
  const int sel = lane >> 4, lm = lane & 15;

  // staging: A tile 128x32 halves = 512 x 16B (2 chunks/thread); B tile same.
  auto stage = [&](int bbuf, int k0) {
#pragma unroll
    for (int i = 0; i < 2; ++i) {
      int q = tid + i * 256;
      int row = q >> 2, c8 = (q & 3) * 8;
      int rg = m0 + row; if (rg >= M) rg = M - 1;
      async_copy16(A + (size_t)rg * K + k0 + c8, &As[bbuf][row * 32 + c8]);
    }
#pragma unroll
    for (int i = 0; i < 2; ++i) {
      int q = tid + i * 256;
      int col = q >> 2, c8 = (q & 3) * 8;
      async_copy16(Bt + (size_t)(n0 + col) * K + k0 + c8, &Bs[bbuf][col * 32 + c8]);
    }
  };

  v8f acc[8] = {};
  stage(0, 0);
  async_wait();
  __syncthreads();
  int buf = 0;
  for (int k0 = 0; k0 < K; k0 += 32) {
    if (k0 + 32 < K) stage(buf ^ 1, k0 + 32);
    // batch all LDS fragment loads first, then run the WMMA chain
    F16x16 a;
    const half_t* ar = &As[buf][(wave * 16 + lm) * 32];
    a.v = *(const v16h*)(ar + sel * 16);
    v16h bfrag[8];
#pragma unroll
    for (int t = 0; t < 8; ++t)
      bfrag[t] = *(const v16h*)(&Bs[buf][(t * 16 + lm) * 32 + sel * 16]);
#pragma unroll
    for (int t = 0; t < 8; ++t)
      acc[t] = __builtin_amdgcn_wmma_f32_16x16x32_f16(false, a.v, false, bfrag[t],
                                                      (short)0, acc[t], false, false);
    async_wait();
    __syncthreads();
    buf ^= 1;
  }

  const int mw = m0 + wave * 16;       // this wave's row block (wave-uniform guard)
  if (mw < M) {
#pragma unroll
    for (int t = 0; t < 8; ++t) {
      int col = n0 + t * 16 + lm;
      float bv = bias ? bias[col] : 0.f;
#pragma unroll
      for (int r = 0; r < 8; ++r) {
        int row = mw + sel * 8 + r;
        float v = acc[t][r] + bv;
        if (act == 1) v = 0.5f * v * (1.f + erff(v * 0.70710678118654752f));
        int orow = outRows ? outRows[row] : row;
        size_t off = (size_t)orow * N + col;
        if (res) v += res[off];
        if (outF) outF[off] = v;
        else      outH[off] = (half_t)v;
      }
    }
  }
}

// ---------------------------------------------------------------- flash attention
// Qh,Kh: (Bq*NH, NTOK, HD) f16 ; Vt: (Bq*NH, HD, NTOK) f16 ; ctxo: (Bq, NTOK, DMODEL) f16
__global__ __launch_bounds__(128)
void attn_kernel(const half_t* __restrict__ Qh, const half_t* __restrict__ Kh,
                 const half_t* __restrict__ Vt, half_t* __restrict__ ctxo) {
  __shared__ __align__(32) half_t pl[4][16 * 32];
  int lane = threadIdx.x & 31, wave = threadIdx.x >> 5;
  int bh = blockIdx.y;
  int q0 = (blockIdx.x * 4 + wave) * 16;
  if (q0 >= NTOK) return;
  int sel = lane >> 4, lm = lane & 15;
  int b = bh / NHEAD, hh = bh % NHEAD;
  const half_t* Qb = Qh + (size_t)bh * NTOK * HD;
  const half_t* Kb = Kh + (size_t)bh * NTOK * HD;
  const half_t* Vb = Vt + (size_t)bh * HD * NTOK;

  F16x16 qa;
  const half_t* qrow = Qb + (size_t)(q0 + lm) * HD;
  qa.v = *(const v16h*)(qrow + sel * 16);

  float mrow[8], lrow[8];
  v8f o0 = {}, o1 = {};
#pragma unroll
  for (int r = 0; r < 8; ++r) { mrow[r] = -1e30f; lrow[r] = 0.f; }

  for (int kb = 0; kb < NTOK; kb += 32) {
    v16h bk0 = *(const v16h*)(Kb + (size_t)(kb + lm) * HD + sel * 16);
    v16h bk1 = *(const v16h*)(Kb + (size_t)(kb + 16 + lm) * HD + sel * 16);
    // prefetch V fragments early so they overlap the softmax VALU work
    v16h bv0 = *(const v16h*)(Vb + (size_t)lm * NTOK + kb + sel * 16);
    v16h bv1 = *(const v16h*)(Vb + (size_t)(16 + lm) * NTOK + kb + sel * 16);
    v8f z = {};
    v8f s0 = __builtin_amdgcn_wmma_f32_16x16x32_f16(false, qa.v, false, bk0, (short)0, z, false, false);
    v8f s1 = __builtin_amdgcn_wmma_f32_16x16x32_f16(false, qa.v, false, bk1, (short)0, z, false, false);
#pragma unroll
    for (int r = 0; r < 8; ++r) {
      float a0 = s0[r] * ATT_SCALE, a1 = s1[r] * ATT_SCALE;
      float mx = fmaxf(a0, a1);
      for (int d = 1; d < 16; d <<= 1) mx = fmaxf(mx, __shfl_xor(mx, d, 32));
      float nm = fmaxf(mrow[r], mx);
      float p0 = __expf(a0 - nm), p1 = __expf(a1 - nm);
      float sm = p0 + p1;
      for (int d = 1; d < 16; d <<= 1) sm += __shfl_xor(sm, d, 32);
      float alpha = __expf(mrow[r] - nm);
      lrow[r] = lrow[r] * alpha + sm;
      o0[r] *= alpha; o1[r] *= alpha;
      mrow[r] = nm;
      pl[wave][(sel * 8 + r) * 32 + lm]      = (half_t)p0;
      pl[wave][(sel * 8 + r) * 32 + 16 + lm] = (half_t)p1;
    }
    lds_wait();
    F16x16 pa;
    pa.v = *(const v16h*)(&pl[wave][lm * 32 + sel * 16]);
    o0 = __builtin_amdgcn_wmma_f32_16x16x32_f16(false, pa.v, false, bv0, (short)0, o0, false, false);
    o1 = __builtin_amdgcn_wmma_f32_16x16x32_f16(false, pa.v, false, bv1, (short)0, o1, false, false);
  }
#pragma unroll
  for (int r = 0; r < 8; ++r) {
    int tok = q0 + sel * 8 + r;
    float inv = 1.f / lrow[r];
    size_t base = ((size_t)(b * NTOK + tok)) * DMODEL + hh * HD;
    ctxo[base + lm]      = (half_t)(o0[r] * inv);
    ctxo[base + 16 + lm] = (half_t)(o1[r] * inv);
  }
}

// ---------------------------------------------------------------- host orchestration
extern "C" void kernel_launch(void* const* d_in, const int* in_sizes, int n_in,
                              void* d_out, int out_size, void* d_ws, size_t ws_size,
                              hipStream_t stream) {
  (void)in_sizes; (void)n_in; (void)out_size; (void)ws_size;
  const float* enc        = (const float*)d_in[0];
  const int*   cm         = (const int*)d_in[1];
  const int*   tm         = (const int*)d_in[2];
  const float* embed_w    = (const float*)d_in[3];
  const float* embed_b    = (const float*)d_in[4];
  const float* mask_token = (const float*)d_in[5];
  const float* ln_g       = (const float*)d_in[198];
  const float* ln_b       = (const float*)d_in[199];
  const float* proj_w     = (const float*)d_in[200];
  const float* proj_b     = (const float*)d_in[201];

  size_t off = 0;
  auto alloc = [&](size_t bytes) -> void* {
    void* p = (char*)d_ws + off;
    off += (bytes + 255) & ~(size_t)255;
    return p;
  };

  half_t* wEmbT  = (half_t*)alloc((size_t)DMODEL * ENCD * 2);
  half_t* wqT[NLAYER], *wkT[NLAYER], *wvT[NLAYER], *woT[NLAYER], *wf1T[NLAYER], *wf2T[NLAYER];
  for (int i = 0; i < NLAYER; ++i) {
    wqT[i]  = (half_t*)alloc((size_t)DMODEL * DMODEL * 2);
    wkT[i]  = (half_t*)alloc((size_t)DMODEL * DMODEL * 2);
    wvT[i]  = (half_t*)alloc((size_t)DMODEL * DMODEL * 2);
    woT[i]  = (half_t*)alloc((size_t)DMODEL * DMODEL * 2);
    wf1T[i] = (half_t*)alloc((size_t)MLPD * DMODEL * 2);
    wf2T[i] = (half_t*)alloc((size_t)DMODEL * MLPD * 2);
  }
  half_t* wProjT = (half_t*)alloc((size_t)ENCD * DMODEL * 2);
  float*  h      = (float*)alloc((size_t)TOTROWS * DMODEL * 4);
  half_t* xln    = (half_t*)alloc((size_t)TOTROWS * DMODEL * 2);
  float*  qf     = (float*)alloc((size_t)TOTROWS * DMODEL * 4);
  float*  kf     = (float*)alloc((size_t)TOTROWS * DMODEL * 4);
  float*  vf     = (float*)alloc((size_t)TOTROWS * DMODEL * 4);
  half_t* Qh     = (half_t*)alloc((size_t)TOTROWS * DMODEL * 2);
  half_t* Kh     = (half_t*)alloc((size_t)TOTROWS * DMODEL * 2);
  half_t* Vt     = (half_t*)alloc((size_t)TOTROWS * DMODEL * 2);
  half_t* ctxo   = (half_t*)alloc((size_t)TOTROWS * DMODEL * 2);
  half_t* actb   = (half_t*)alloc((size_t)TOTROWS * MLPD * 2);
  half_t* Actx   = (half_t*)alloc((size_t)Bq * NCTX * ENCD * 2);
  half_t* Afin   = (half_t*)alloc((size_t)Bq * NTGT * DMODEL * 2);
  int*    ctx_rows = (int*)alloc((size_t)Bq * NCTX * 4);
  int*    tgt_rows = (int*)alloc((size_t)Bq * NTGT * 4);

  auto trans = [&](const float* src, half_t* dst, int K, int N) {
    size_t n = (size_t)K * N;
    transpose_to_f16_kernel<<<(unsigned)((n + 255) / 256), 256, 0, stream>>>(src, dst, K, N);
  };
  auto gemm = [&](const half_t* A, const half_t* Bt, const float* bias, const float* res,
                  float* outF, half_t* outH, const int* outRows, int M, int N, int K, int act) {
    dim3 g(N / 128, (M + 127) / 128);
    gemm_f16_kernel<<<g, 256, 0, stream>>>(A, Bt, bias, res, outF, outH, outRows, M, N, K, act);
  };

  // --- weight prep
  trans(embed_w, wEmbT, ENCD, DMODEL);
  for (int i = 0; i < NLAYER; ++i) {
    const int base = 6 + 16 * i;
    trans((const float*)d_in[base + 2],  wqT[i],  DMODEL, DMODEL);
    trans((const float*)d_in[base + 4],  wkT[i],  DMODEL, DMODEL);
    trans((const float*)d_in[base + 6],  wvT[i],  DMODEL, DMODEL);
    trans((const float*)d_in[base + 8],  woT[i],  DMODEL, DMODEL);
    trans((const float*)d_in[base + 12], wf1T[i], DMODEL, MLPD);
    trans((const float*)d_in[base + 14], wf2T[i], MLPD, DMODEL);
  }
  trans(proj_w, wProjT, DMODEL, ENCD);

  // --- token assembly (argsort collapses: pos == arange, scatter by mask value)
  build_rows_kernel<<<(Bq * NCTX + 255) / 256, 256, 0, stream>>>(cm, tm, ctx_rows, tgt_rows);
  gather_enc_kernel<<<Bq * NCTX, 256, 0, stream>>>(enc, cm, Actx);
  init_tgt_kernel<<<Bq * NTGT, 128, 0, stream>>>(tgt_rows, mask_token, h);
  gemm(Actx, wEmbT, embed_b, nullptr, h, nullptr, ctx_rows, Bq * NCTX, DMODEL, ENCD, 0);

  // --- transformer layers
  const size_t rope_n = (size_t)TOTROWS * DMODEL;
  for (int i = 0; i < NLAYER; ++i) {
    const int base = 6 + 16 * i;
    const float* ln1_g = (const float*)d_in[base + 0];
    const float* ln1_b = (const float*)d_in[base + 1];
    const float* q_b   = (const float*)d_in[base + 3];
    const float* k_b   = (const float*)d_in[base + 5];
    const float* v_b   = (const float*)d_in[base + 7];
    const float* o_b   = (const float*)d_in[base + 9];
    const float* ln2_g = (const float*)d_in[base + 10];
    const float* ln2_b = (const float*)d_in[base + 11];
    const float* f1_b  = (const float*)d_in[base + 13];
    const float* f2_b  = (const float*)d_in[base + 15];

    layernorm_kernel<<<(TOTROWS + 3) / 4, 128, 0, stream>>>(h, ln1_g, ln1_b, xln, TOTROWS);
    gemm(xln, wqT[i], q_b, nullptr, qf, nullptr, nullptr, TOTROWS, DMODEL, DMODEL, 0);
    gemm(xln, wkT[i], k_b, nullptr, kf, nullptr, nullptr, TOTROWS, DMODEL, DMODEL, 0);
    gemm(xln, wvT[i], v_b, nullptr, vf, nullptr, nullptr, TOTROWS, DMODEL, DMODEL, 0);
    rope_pack_kernel<<<(unsigned)((rope_n + 255) / 256), 256, 0, stream>>>(qf, kf, vf, Qh, Kh, Vt);
    {
      dim3 g((NTOK / 16 + 3) / 4, Bq * NHEAD);
      attn_kernel<<<g, 128, 0, stream>>>(Qh, Kh, Vt, ctxo);
    }
    gemm(ctxo, woT[i], o_b, h, h, nullptr, nullptr, TOTROWS, DMODEL, DMODEL, 0);
    layernorm_kernel<<<(TOTROWS + 3) / 4, 128, 0, stream>>>(h, ln2_g, ln2_b, xln, TOTROWS);
    gemm(xln, wf1T[i], f1_b, nullptr, nullptr, actb, nullptr, TOTROWS, MLPD, DMODEL, 1);
    gemm(actb, wf2T[i], f2_b, h, h, nullptr, nullptr, TOTROWS, DMODEL, MLPD, 0);
  }

  // --- final LN, gather targets, project
  layernorm_kernel<<<(TOTROWS + 3) / 4, 128, 0, stream>>>(h, ln_g, ln_b, xln, TOTROWS);
  gather_f16_kernel<<<Bq * NTGT, 128, 0, stream>>>(xln, tgt_rows, Afin);
  gemm(Afin, wProjT, proj_b, nullptr, (float*)d_out, nullptr, nullptr,
       Bq * NTGT, ENCD, DMODEL, 0);
}
